// ASA_37692632989803
// MI455X (gfx1250) — compile-verified, matrix operand order checked by reading, not run on hardware
//
#include <hip/hip_runtime.h>
#include <hip/hip_bf16.h>
#include <hip/hip_fp16.h>

typedef float    v2f  __attribute__((ext_vector_type(2)));
typedef float    v8f  __attribute__((ext_vector_type(8)));
typedef _Float16 v16h __attribute__((ext_vector_type(16)));

#define N_  32
#define T_  8
#define C_  128
#define HW_ 1024
#define TC_ (T_*C_)   // 1024
#define KC_ 64
#define KT_ 4

// ---------------------------------------------------------------------------
// Kernel 1: per-(n,t,c) spatial sum & max over 1024 pixels. One wave per row.
// ---------------------------------------------------------------------------
__global__ void k_pool(const float* __restrict__ x,
                       float* __restrict__ psum, float* __restrict__ pmax) {
    int wave = threadIdx.x >> 5, lane = threadIdx.x & 31;
    int row  = blockIdx.x * 8 + wave;                    // 0 .. 32767
    const float4* xr = reinterpret_cast<const float4*>(x) + (size_t)row * 256;
    float s = 0.f, m = -__builtin_inff();
#pragma unroll
    for (int j = 0; j < 8; ++j) {
        float4 v = xr[lane + 32 * j];
        s += v.x + v.y + v.z + v.w;
        m = fmaxf(m, fmaxf(fmaxf(v.x, v.y), fmaxf(v.z, v.w)));
    }
#pragma unroll
    for (int off = 16; off > 0; off >>= 1) {
        s += __shfl_xor(s, off, 32);
        m = fmaxf(m, __shfl_xor(m, off, 32));
    }
    if (lane == 0) { psum[row] = s; pmax[row] = m; }
}

// ---------------------------------------------------------------------------
// Kernel 2: time-FC  y[n,s,c] = sum_t Wfc[s,t] * map_add[n,t,c]  via WMMA.
// One wave per (n, 16-channel chunk); 16x16 tile, two K=4 steps (t=0..7).
// fp32 WMMA => bit-faithful fp32 matmul.
// ---------------------------------------------------------------------------
__global__ void k_fc(const float* __restrict__ psum, const float* __restrict__ pmax,
                     const float* __restrict__ Wfc,  const float* __restrict__ alpha,
                     const float* __restrict__ beta, float* __restrict__ mfc) {
    int n    = blockIdx.x;
    int wave = threadIdx.x >> 5, lane = threadIdx.x & 31;
    int c0   = wave * 16;
    int m    = lane & 15;          // A: row s / B: col c
    int khi  = lane >> 4;          // K half select
    float ca = 0.5f + alpha[0], cb = 0.5f + beta[0];
    const float inv = 1.0f / 1024.0f;
    v8f acc = {};
#if __has_builtin(__builtin_amdgcn_wmma_f32_16x16x4_f32)
#pragma unroll
    for (int kb = 0; kb < 8; kb += 4) {
        int t0 = kb + 2 * khi;                    // K index of vgpr 0 (<= 7)
        v2f a, b;
        a.x = (m < 8) ? Wfc[m * 8 + t0]     : 0.f;   // A[m][k] = Wfc[s,t]
        a.y = (m < 8) ? Wfc[m * 8 + t0 + 1] : 0.f;
        int i0 = (n * T_ + t0) * C_ + c0 + m;        // B[k][n] = map_add[t,c]
        b.x = ca * psum[i0] * inv + cb * pmax[i0];
        int i1 = i0 + C_;
        b.y = ca * psum[i1] * inv + cb * pmax[i1];
        acc = __builtin_amdgcn_wmma_f32_16x16x4_f32(false, a, false, b,
                                                    (short)0, acc, false, false);
    }
#else
    // Fallback: f16 WMMA (t=8 zero-padded to K=32).
    v16h A = {}, B = {};
#pragma unroll
    for (int hv = 0; hv < 16; ++hv) {
        int vg = hv >> 1, sl = hv & 1;
        int k = (vg < 4) ? (khi * 8 + vg * 2 + sl) : (16 + khi * 8 + (vg - 4) * 2 + sl);
        if (k < 8 && m < 8) A[hv] = (_Float16)Wfc[m * 8 + k];
        int kb2 = khi * 16 + hv;
        if (kb2 < 8) {
            int ii = (n * T_ + kb2) * C_ + c0 + m;
            B[hv] = (_Float16)(ca * psum[ii] * inv + cb * pmax[ii]);
        }
    }
    acc = __builtin_amdgcn_wmma_f32_16x16x32_f16(false, A, false, B,
                                                 (short)0, acc, false, false);
#endif
    if (lane < 16) {
#pragma unroll
        for (int r = 0; r < 8; ++r)
            mfc[(n * T_ + r) * C_ + c0 + lane] = acc[r];   // rows s=0..7
    }
}

// ---------------------------------------------------------------------------
// Kernel 3a: top-64 channels per (n,t), union over n (rank = #greater + ties<)
// ---------------------------------------------------------------------------
__global__ void k_maskc(const float* __restrict__ mfc, int* __restrict__ mask_c) {
    int nt = blockIdx.x;           // n*8 + t
    int c  = threadIdx.x;          // 0..127
    __shared__ float sv[C_];
    sv[c] = mfc[nt * C_ + c];
    __syncthreads();
    float v = sv[c];
    int rank = 0;
    for (int c2 = 0; c2 < C_; ++c2) {
        float u = sv[c2];
        rank += (u > v) || ((u == v) && (c2 < c));
    }
    if (rank < KC_) atomicOr(&mask_c[(nt & 7) * C_ + c], 1);
}

// ---------------------------------------------------------------------------
// Kernel 3b: top-4 timesteps per (n,c), union over n.
// ---------------------------------------------------------------------------
__global__ void k_maskt(const float* __restrict__ mfc, int* __restrict__ mask_t) {
    int id = blockIdx.x * blockDim.x + threadIdx.x;  // 0..4095
    int n = id >> 7, c = id & 127;
    float vt[T_];
#pragma unroll
    for (int t = 0; t < T_; ++t) vt[t] = mfc[(n * T_ + t) * C_ + c];
#pragma unroll
    for (int t = 0; t < T_; ++t) {
        int rank = 0;
#pragma unroll
        for (int t2 = 0; t2 < T_; ++t2)
            rank += (vt[t2] > vt[t]) || ((vt[t2] == vt[t]) && (t2 < t));
        if (rank < KT_) atomicOr(&mask_t[t * C_ + c], 1);
    }
}

__global__ void k_zero(int* __restrict__ p, int nInt) {
    int i = blockIdx.x * blockDim.x + threadIdx.x;
    if (i < nInt) p[i] = 0;
}

__global__ void k_xor(const int* __restrict__ a, const int* __restrict__ b,
                      int* __restrict__ o) {
    int i = blockIdx.x * blockDim.x + threadIdx.x;
    o[i] = a[i] ^ b[i];
}

// ---------------------------------------------------------------------------
// Kernel 4: per-(n,pixel) masked sum/max over all 1024 (t,c) channels.
// Masked-out entries contribute exact 0 to both sum and max (as in reference,
// where imf = x*mask and the max runs over all channels).
// ---------------------------------------------------------------------------
__global__ void k_stats(const float* __restrict__ x, const int* __restrict__ imtc,
                        float* __restrict__ iavg, float* __restrict__ imax,
                        float* __restrict__ savg, float* __restrict__ smax) {
    int n = blockIdx.x >> 2, pc = blockIdx.x & 3;
    int p = pc * 256 + threadIdx.x;
    const float* xb = x + (size_t)n * TC_ * HW_ + p;
    float is = 0.f, ss = 0.f, im = -__builtin_inff(), sm = -__builtin_inff();
#pragma unroll 4
    for (int tc = 0; tc < TC_; ++tc) {
        float v  = xb[(size_t)tc * HW_];     // coalesced across threads
        int   mk = imtc[tc];                 // uniform -> scalar load
        float vi = mk ? v : 0.f;
        float vs = mk ? 0.f : v;
        is += vi; ss += vs;
        im = fmaxf(im, vi); sm = fmaxf(sm, vs);
    }
    int o = n * HW_ + p;
    iavg[o] = is * (1.f / 512.f);            // mean/1024 then /LAM(=0.5)
    savg[o] = ss * (1.f / 512.f);            // mean/1024 then /(1-LAM)
    imax[o] = im; smax[o] = sm;
}

// ---------------------------------------------------------------------------
// Kernel 5: 3x3 SAME conv on the 2-channel [avg,max] maps + sigmoid.
// ---------------------------------------------------------------------------
__global__ void k_conv(const float* __restrict__ iavg, const float* __restrict__ imax,
                       const float* __restrict__ savg, const float* __restrict__ smax,
                       const float* __restrict__ W1,   const float* __restrict__ W2,
                       float* __restrict__ imap, float* __restrict__ smap) {
    int id = blockIdx.x * blockDim.x + threadIdx.x;   // 0..32767 = n*1024+p
    int n = id >> 10, p = id & 1023;
    int y = p >> 5, xx = p & 31;
    float aI = 0.f, aS = 0.f;
#pragma unroll
    for (int ky = 0; ky < 3; ++ky) {
        int yy = y + ky - 1;
        if (yy < 0 || yy > 31) continue;
#pragma unroll
        for (int kx = 0; kx < 3; ++kx) {
            int xw = xx + kx - 1;
            if (xw < 0 || xw > 31) continue;
            int o = n * HW_ + yy * 32 + xw;
            float w1a = W1[ky * 3 + kx], w1m = W1[9 + ky * 3 + kx];
            float w2a = W2[ky * 3 + kx], w2m = W2[9 + ky * 3 + kx];
            aI += w1a * iavg[o] + w1m * imax[o];
            aS += w2a * savg[o] + w2m * smax[o];
        }
    }
    imap[id] = 1.f / (1.f + expf(-aI));
    smap[id] = 1.f / (1.f + expf(-aS));
}

// ---------------------------------------------------------------------------
// Kernel 6: out = x * (im_tc ? im_map : sub_map). One block per (n,tc) row;
// mask is uniform per block -> scalar branch; float4 I/O.
// ---------------------------------------------------------------------------
__global__ void k_final(const float* __restrict__ x, const int* __restrict__ imtc,
                        const float* __restrict__ imap, const float* __restrict__ smap,
                        float* __restrict__ out) {
    int b  = blockIdx.x;               // n*1024 + tc
    int n  = b >> 10, tc = b & 1023;
    int mk = imtc[tc];                 // uniform
    const float* mp = mk ? imap : smap;
    const float4* x4 = reinterpret_cast<const float4*>(x) + (size_t)b * 256 + threadIdx.x;
    const float4* m4 = reinterpret_cast<const float4*>(mp) + n * 256 + threadIdx.x;
    float4* o4 = reinterpret_cast<float4*>(out) + (size_t)b * 256 + threadIdx.x;
    float4 xv = *x4, mv = *m4, r;
    r.x = xv.x * mv.x; r.y = xv.y * mv.y; r.z = xv.z * mv.z; r.w = xv.w * mv.w;
    *o4 = r;
}

// ---------------------------------------------------------------------------
extern "C" void kernel_launch(void* const* d_in, const int* in_sizes, int n_in,
                              void* d_out, int out_size, void* d_ws, size_t ws_size,
                              hipStream_t stream) {
    const float* x     = (const float*)d_in[0];
    const float* alpha = (const float*)d_in[1];
    const float* beta  = (const float*)d_in[2];
    const float* Wfc   = (const float*)d_in[3];
    const float* W1    = (const float*)d_in[4];
    const float* W2    = (const float*)d_in[5];
    float* out = (float*)d_out;

    float* ws   = (float*)d_ws;            // ~1.2 MB of scratch used
    float* psum = ws;                      // [32768]
    float* pmax = ws + 32768;              // [32768]
    float* mfc  = ws + 65536;              // [32768]
    int*   maskc = (int*)(ws + 98304);     // [1024]
    int*   maskt = (int*)(ws + 99328);     // [1024]  (contiguous with maskc)
    int*   imtc  = (int*)(ws + 100352);    // [1024]
    float* iavg = ws + 101376;             // [32768] each below
    float* imax = ws + 134144;
    float* savg = ws + 166912;
    float* smax = ws + 199680;
    float* imap = ws + 232448;
    float* smap = ws + 265216;

    k_zero <<<8,     256, 0, stream>>>(maskc, 2048);           // maskc+maskt
    k_pool <<<4096,  256, 0, stream>>>(x, psum, pmax);
    k_fc   <<<32,    256, 0, stream>>>(psum, pmax, Wfc, alpha, beta, mfc);
    k_maskc<<<256,   128, 0, stream>>>(mfc, maskc);
    k_maskt<<<16,    256, 0, stream>>>(mfc, maskt);
    k_xor  <<<4,     256, 0, stream>>>(maskc, maskt, imtc);
    k_stats<<<128,   256, 0, stream>>>(x, imtc, iavg, imax, savg, smax);
    k_conv <<<128,   256, 0, stream>>>(iavg, imax, savg, smax, W1, W2, imap, smap);
    k_final<<<32768, 256, 0, stream>>>(x, imtc, imap, smap, out);
}
